// ChebConv_12945031430692
// MI455X (gfx1250) — matrix operand order repaired and checked by reading, not compiled, not run on hardware
//
#include <hip/hip_runtime.h>
#include <hip/hip_bf16.h>

typedef __attribute__((ext_vector_type(16))) __bf16 v16bf;
typedef __attribute__((ext_vector_type(8)))  float  v8f;
typedef __attribute__((__vector_size__(16))) int    i32x4_;

#define NV   20000          // vertices
#define BB_  8              // batch
#define FF   64             // in channels
#define KK   6              // Chebyshev order
#define OUTC 64             // out channels
#define NNZ_ 320000
#define BF   (BB_*FF)       // 512 row width of T buffers
#define ROWS (NV*BB_)       // 160000 GEMM rows
#define KW   (KK*FF)        // 384
#define TELEM ((size_t)NV*BF)   // floats per T buffer

// Native bf16 converts (lower to v_cvt_*bf16_f32 on gfx1250)
__device__ __forceinline__ __bf16 f2bf(float f) { return (__bf16)f; }
__device__ __forceinline__ unsigned short f2bf_bits(float f) {
    __bf16 h = (__bf16)f;
    unsigned short u; __builtin_memcpy(&u, &h, 2); return u;
}

// ---- x [B,N,F] -> T0 [N, B*F]  (float4 granularity) -----------------------
__global__ __launch_bounds__(256) void k_transpose(const float* __restrict__ x,
                                                   float* __restrict__ t0) {
    int tid = blockIdx.x * 256 + threadIdx.x;        // NV*128 threads exactly
    int c4 = tid & 127;                              // b*16 + f4
    int n  = tid >> 7;
    int b  = c4 >> 4, f4 = c4 & 15;
    float4 v = reinterpret_cast<const float4*>(x)[(size_t)b * NV * 16 + (size_t)n * 16 + f4];
    reinterpret_cast<float4*>(t0)[(size_t)n * 128 + c4] = v;
}

// ---- dst = scale * dst (scale=0 zeroes, scale=-1 negates in place) --------
__global__ __launch_bounds__(256) void k_scale(float* __restrict__ dst, float scale) {
    int tid = blockIdx.x * 256 + threadIdx.x;        // TELEM/4 threads exactly
    float4* p = reinterpret_cast<float4*>(dst);
    float4 v = p[tid];
    v.x *= scale; v.y *= scale; v.z *= scale; v.w *= scale;
    p[tid] = v;
}

__global__ __launch_bounds__(256) void k_izero(int* __restrict__ p, int n) {
    int i = blockIdx.x * 256 + threadIdx.x;
    if (i < n) p[i] = 0;
}

__global__ __launch_bounds__(256) void k_count(const int* __restrict__ rows,
                                               int* __restrict__ cnt, int nnz) {
    int e = blockIdx.x * 256 + threadIdx.x;
    if (e < nnz) atomicAdd(&cnt[rows[e]], 1);
}

// single-workgroup exclusive scan of per-row counts -> rowptr, cursor
__global__ __launch_bounds__(256) void k_scan(const int* __restrict__ cnt,
                                              int* __restrict__ rowptr,
                                              int* __restrict__ cursor, int n) {
    __shared__ int s[256];
    int t = threadIdx.x;
    const int CH = (NV + 255) / 256;                 // 79
    int base = t * CH, sum = 0;
    for (int i = 0; i < CH; ++i) { int idx = base + i; if (idx < n) sum += cnt[idx]; }
    s[t] = sum; __syncthreads();
    for (int off = 1; off < 256; off <<= 1) {        // inclusive Hillis-Steele
        int v = (t >= off) ? s[t - off] : 0;
        __syncthreads(); s[t] += v; __syncthreads();
    }
    int run = s[t] - sum;                            // exclusive prefix
    for (int i = 0; i < CH; ++i) {
        int idx = base + i;
        if (idx < n) { rowptr[idx] = run; cursor[idx] = run; run += cnt[idx]; }
    }
    if (t == 255) rowptr[n] = s[255];
}

__global__ __launch_bounds__(256) void k_scatter(const int* __restrict__ rows,
                                                 const int* __restrict__ cols,
                                                 const float* __restrict__ vals,
                                                 int* __restrict__ cursor,
                                                 int* __restrict__ ccols,
                                                 float* __restrict__ cvals, int nnz) {
    int e = blockIdx.x * 256 + threadIdx.x;
    if (e < nnz) {
        int p = atomicAdd(&cursor[rows[e]], 1);
        ccols[p] = cols[e]; cvals[p] = vals[e];
    }
}

// ---- race-free CSR SpMM: dst[row,:] += alpha * sum val * src[col,:] -------
__global__ __launch_bounds__(128) void k_spmm(const int* __restrict__ rowptr,
                                              const int* __restrict__ ccols,
                                              const float* __restrict__ cvals,
                                              const float* __restrict__ src,
                                              float* __restrict__ dst, float alpha) {
    int row = blockIdx.x;
    int ch4 = threadIdx.x;                           // 128 float4 lanes = 512 ch
    int s = rowptr[row], e = rowptr[row + 1];
    float4 acc = make_float4(0.f, 0.f, 0.f, 0.f);
    for (int i = s; i < e; ++i) {
        int   c = ccols[i];
        float v = cvals[i];
        float4 g = reinterpret_cast<const float4*>(src + (size_t)c * BF)[ch4];
        acc.x += v * g.x; acc.y += v * g.y; acc.z += v * g.z; acc.w += v * g.w;
    }
    float4* d = reinterpret_cast<float4*>(dst + (size_t)row * BF);
    float4 cur = d[ch4];
    cur.x += alpha * acc.x; cur.y += alpha * acc.y;
    cur.z += alpha * acc.z; cur.w += alpha * acc.w;
    d[ch4] = cur;
}

// ---- W [OUT, K*F] -> bf16 WMMA B-fragments, layout [kc][kb][ct][lane][16] --
__global__ __launch_bounds__(256) void k_wconvert(const float* __restrict__ W,
                                                  unsigned short* __restrict__ wfrag) {
    int e = blockIdx.x * 256 + threadIdx.x;          // 24576 exactly
    int j  = e & 15, lane = (e >> 4) & 31, ct = (e >> 9) & 3;
    int kb = (e >> 11) & 1, kc = e >> 12;
    int hi = lane >> 4;
    int o  = ct * 16 + (lane & 15);                  // output column
    int kin = kb * 32 + 8 * hi + j + (j >= 8 ? 8 : 0); // B 32x16 bf16 layout
    int fullk = kc * 64 + kin;
    wfrag[e] = f2bf_bits(W[o * KW + fullk]);
}

// ---- accumulating bf16 WMMA GEMM pass: out += T_kc @ W_kc^T (+bias) -------
__global__ __launch_bounds__(256) void k_gemm(const float* __restrict__ Tsrc,
                                              const unsigned short* __restrict__ wfrag,
                                              const float* __restrict__ bias,
                                              float* __restrict__ out,
                                              int kc, int first) {
    __shared__ unsigned short sB[2 * 4 * 32 * 16];   // 8 KB of B fragments
    int tid = threadIdx.x;

    // stage this pass's W fragments into LDS (8 KB)
#if __has_builtin(__builtin_amdgcn_global_load_async_to_lds_b128) && \
    __has_builtin(__builtin_amdgcn_s_wait_asynccnt)
    {
        __attribute__((address_space(1))) char* g =
            (__attribute__((address_space(1))) char*)(wfrag + (size_t)kc * 4096);
        __attribute__((address_space(3))) char* l =
            (__attribute__((address_space(3))) char*)sB;
        __builtin_amdgcn_global_load_async_to_lds_b128(
            (__attribute__((address_space(1))) i32x4_*)(g + tid * 16),
            (__attribute__((address_space(3))) i32x4_*)(l + tid * 16), 0, 0);
        __builtin_amdgcn_global_load_async_to_lds_b128(
            (__attribute__((address_space(1))) i32x4_*)(g + 4096 + tid * 16),
            (__attribute__((address_space(3))) i32x4_*)(l + 4096 + tid * 16), 0, 0);
        __builtin_amdgcn_s_wait_asynccnt(0);
    }
#else
    {
        const uint4* g = reinterpret_cast<const uint4*>(wfrag + (size_t)kc * 4096);
        uint4* l = reinterpret_cast<uint4*>(sB);
        l[tid] = g[tid];
        l[tid + 256] = g[tid + 256];
    }
#endif
    __syncthreads();

    int lane = tid & 31, wv = tid >> 5;
    int hi = lane >> 4, mrow = lane & 15;
    int rbase = blockIdx.x * 128 + wv * 16;
    int r = rbase + mrow;
    int n = r >> 3, b = r & 7;
    const float* arow = Tsrc + (size_t)n * BF + (size_t)b * FF;

    // A fragments for kb = 0,1 (16x32 bf16 layout)
    v16bf afr[2];
#pragma unroll
    for (int kb = 0; kb < 2; ++kb) {
        int seg0 = kb * 32 + 8 * hi;
        float4 f0 = *reinterpret_cast<const float4*>(arow + seg0);
        float4 f1 = *reinterpret_cast<const float4*>(arow + seg0 + 4);
        float4 f2 = *reinterpret_cast<const float4*>(arow + seg0 + 16);
        float4 f3 = *reinterpret_cast<const float4*>(arow + seg0 + 20);
        v16bf a;
        a[0]=f2bf(f0.x);  a[1]=f2bf(f0.y);  a[2]=f2bf(f0.z);  a[3]=f2bf(f0.w);
        a[4]=f2bf(f1.x);  a[5]=f2bf(f1.y);  a[6]=f2bf(f1.z);  a[7]=f2bf(f1.w);
        a[8]=f2bf(f2.x);  a[9]=f2bf(f2.y);  a[10]=f2bf(f2.z); a[11]=f2bf(f2.w);
        a[12]=f2bf(f3.x); a[13]=f2bf(f3.y); a[14]=f2bf(f3.z); a[15]=f2bf(f3.w);
        afr[kb] = a;
    }

    int ocol_lo = lane & 15;
#pragma unroll
    for (int ct = 0; ct < 4; ++ct) {
        int ocol = ct * 16 + ocol_lo;
        v8f c;
        if (first) {
            float bv = bias[ocol];
#pragma unroll
            for (int j = 0; j < 8; ++j) c[j] = bv;
        } else {
#pragma unroll
            for (int j = 0; j < 8; ++j) {           // C/D layout: M = j + 8*hi
                int rr = rbase + j + 8 * hi;
                c[j] = out[(size_t)(rr & 7) * (NV * OUTC) + (size_t)(rr >> 3) * OUTC + ocol];
            }
        }
#pragma unroll
        for (int kb = 0; kb < 2; ++kb) {
            v16bf bf = *reinterpret_cast<const v16bf*>(&sB[((kb * 4 + ct) * 32 + lane) * 16]);
            c = __builtin_amdgcn_wmma_f32_16x16x32_bf16(false, afr[kb], false, bf,
                                                        (short)0, c, false, false);
        }
#pragma unroll
        for (int j = 0; j < 8; ++j) {               // fused [N*B,OUT]->[B,N,OUT]
            int rr = rbase + j + 8 * hi;
            out[(size_t)(rr & 7) * (NV * OUTC) + (size_t)(rr >> 3) * OUTC + ocol] = c[j];
        }
    }
}

extern "C" void kernel_launch(void* const* d_in, const int* in_sizes, int n_in,
                              void* d_out, int out_size, void* d_ws, size_t ws_size,
                              hipStream_t stream) {
    const float* x    = (const float*)d_in[0];
    const int*   rows = (const int*)d_in[1];
    const int*   cols = (const int*)d_in[2];
    const float* vals = (const float*)d_in[3];
    const float* W    = (const float*)d_in[4];
    const float* bias = (const float*)d_in[5];
    float* out = (float*)d_out;

    char* p = (char*)d_ws;
    auto bump = [&](size_t bytes) -> void* {
        void* r = (void*)p; p += (bytes + 255) & ~((size_t)255); return r;
    };
    float* bufA = (float*)bump(TELEM * 4);           // T_{even}
    float* bufB = (float*)bump(TELEM * 4);           // T_{odd}
    unsigned short* wfrag = (unsigned short*)bump(KK * 4096 * 2);
    int*   cnt    = (int*)bump(NV * 4);
    int*   rowptr = (int*)bump((NV + 1) * 4);
    int*   cursor = (int*)bump(NV * 4);
    int*   ccols  = (int*)bump(NNZ_ * 4);
    float* cvals  = (float*)bump(NNZ_ * 4);

    const int gScale = (int)(TELEM / 4 / 256);       // 10000
    const int gTr    = NV * 128 / 256;               // 10000
    const int gNnz   = (NNZ_ + 255) / 256;
    const int gGemm  = ROWS / 128;                   // 1250

    // W -> bf16 fragments; x -> T0
    k_wconvert<<<KK * 4096 / 256, 256, 0, stream>>>(W, wfrag);
    k_transpose<<<gTr, 256, 0, stream>>>(x, bufA);

    // build row-sorted CSR copy of the COO Laplacian
    k_izero<<<(NV + 255) / 256, 256, 0, stream>>>(cnt, NV);
    k_count<<<gNnz, 256, 0, stream>>>(rows, cnt, NNZ_);
    k_scan<<<1, 256, 0, stream>>>(cnt, rowptr, cursor, NV);
    k_scatter<<<gNnz, 256, 0, stream>>>(rows, cols, vals, cursor, ccols, cvals, NNZ_);

    // k = 0 : out = bias + T0 @ W0^T
    k_gemm<<<gGemm, 256, 0, stream>>>(bufA, wfrag, bias, out, 0, 1);

    // k = 1 : T1 = L T0
    k_scale<<<gScale, 256, 0, stream>>>(bufB, 0.f);
    k_spmm<<<NV, 128, 0, stream>>>(rowptr, ccols, cvals, bufA, bufB, 1.f);
    k_gemm<<<gGemm, 256, 0, stream>>>(bufB, wfrag, bias, out, 1, 0);

    // k = 2..5 : T_k = 2 L T_{k-1} - T_{k-2}, overwriting T_{k-2} in place
    for (int k = 2; k < KK; ++k) {
        float* dst = (k & 1) ? bufB : bufA;
        float* src = (k & 1) ? bufA : bufB;
        k_scale<<<gScale, 256, 0, stream>>>(dst, -1.f);
        k_spmm<<<NV, 128, 0, stream>>>(rowptr, ccols, cvals, src, dst, 2.f);
        k_gemm<<<gGemm, 256, 0, stream>>>(dst, wfrag, bias, out, k, 0);
    }
}
